// Gemma4AudioExportModel_4939212390919
// MI455X (gfx1250) — compile-verified, matrix operand order checked by reading, not run on hardware
//
#include <hip/hip_runtime.h>
#include <hip/hip_bf16.h>
#include <math.h>

typedef __attribute__((ext_vector_type(16))) _Float16 v16h;
typedef __attribute__((ext_vector_type(8)))  float    v8f;
typedef __attribute__((ext_vector_type(4)))  unsigned int v4u;
typedef __attribute__((ext_vector_type(8)))  int v8i;
typedef __attribute__((ext_vector_type(4)))  int v4i;

constexpr int kB    = 4;
constexpr int kHID  = 1536;
constexpr int kH    = 8;
constexpr int kD    = 192;
constexpr int kCHUNK = 12;
constexpr int kCTX   = 24;
constexpr int kPL    = 13;

// ---------------------------------------------------------------- WMMA helpers
__device__ __forceinline__ v8f wmma16(v16h a, v16h b, v8f c) {
  return __builtin_amdgcn_wmma_f32_16x16x32_f16(false, a, false, b, (short)0, c,
                                                false, false);
}

__device__ __forceinline__ v16h frag_zero() {
  union { unsigned u[8]; v16h h; } t;
#pragma unroll
  for (int i = 0; i < 8; ++i) t.u[i] = 0u;
  return t.h;
}

// Fragment = 16 halves at p[0..7] and p[16..23] (A/B f16 lane layout).
__device__ __forceinline__ v16h frag_ld(const _Float16* p) {
  union { v4u u[2]; v16h h; } t;
  t.u[0] = *(const v4u*)(p);
  t.u[1] = *(const v4u*)(p + 16);
  return t.h;
}

// ------------------------------------------------- Tensor Data Mover (2D tile)
// Packs D# group0/group1 per CDNA5 ISA and issues
// tensor_load_to_lds (6-arg clang-23 builtin). data_size = 2 bytes.
// pad_on: insert 16B LDS pad every 64B (row pitch 80B -> bank-conflict-free).
__device__ __forceinline__ void tdm_load_2d(unsigned lds_off, const void* gsrc,
                                            unsigned td0, unsigned td1,
                                            unsigned tile0, unsigned tile1,
                                            unsigned long long stride0,
                                            int pad_on) {
  unsigned long long ga = (unsigned long long)gsrc;
  v4u g0;
  g0.x = 1u;                                   // count=1, user mode, no gather
  g0.y = lds_off;                              // LDS byte address
  g0.z = (unsigned)ga;                         // global_addr[31:0]
  g0.w = (unsigned)((ga >> 32) & 0x1FFFFFFull) | (2u << 30);  // addr[56:32],type=2
  unsigned d0 = 1u << 16;                      // data_size=1 -> 2 bytes
  if (pad_on) d0 |= (1u << 20) | (3u << 22) | (3u << 25);  // pad: every 16dw, +4dw
  v8i g1;
  g1[0] = (int)d0;
  g1[1] = (int)((td0 & 0xFFFFu) << 16);        // tensor_dim0 lo16 (atomic addr 0)
  g1[2] = (int)((td0 >> 16) | ((td1 & 0xFFFFu) << 16));
  g1[3] = (int)((td1 >> 16) | (tile0 << 16));  // tile_dim0
  g1[4] = (int)(tile1 & 0xFFFFu);              // tile_dim1 (tile_dim2 = 0)
  g1[5] = (int)(unsigned)stride0;              // tensor_dim0_stride lo32
  g1[6] = (int)(unsigned)((stride0 >> 32) & 0xFFFFull);
  g1[7] = 0;
  v4i z4; z4[0] = z4[1] = z4[2] = z4[3] = 0;   // groups 2/3 unused (2D tensor)
  v8i z8;
#pragma unroll
  for (int i = 0; i < 8; ++i) z8[i] = 0;
  __builtin_amdgcn_tensor_load_to_lds(g0, g1, z4, z4, z8, 0);
}

// -------------------------------------------------------- f16 WMMA tiled GEMM
// C[M,N] = A[M,K] * Bt[N,K]^T ; A,Bt row-major f16 (K contiguous).
// Per-column scale optional; output f32 or f16.
constexpr int BM = 128, BN = 128, BK = 32;
constexpr int AROW = 40;                      // halves/row in LDS (32 + 8 pad)
constexpr int STAGE_H = (BM + BN) * AROW;     // halves per stage
constexpr int STAGE_BYTES = STAGE_H * 2;

__global__ __launch_bounds__(256) void gemm_f16_wmma(
    const _Float16* __restrict__ A, const _Float16* __restrict__ Bt,
    void* __restrict__ C, const float* __restrict__ colscale,
    int M, int N, int K, int c_f32) {
  __shared__ _Float16 sm[2 * STAGE_H];
  const int tid = threadIdx.x;
  const int lane = tid & 31, w = tid >> 5;
  const int wm = w & 3, wn = w >> 2;
  const int hl = lane >> 4, r = lane & 15;
  const int row0 = blockIdx.y * BM, col0 = blockIdx.x * BN;

  const unsigned lds0 = (unsigned)(unsigned long long)(void*)&sm[0];

  v8f acc[2][4];
#pragma unroll
  for (int i = 0; i < 2; ++i)
#pragma unroll
    for (int j = 0; j < 4; ++j)
#pragma unroll
      for (int e = 0; e < 8; ++e) acc[i][j][e] = 0.0f;

  const int nk = K / BK;
  if (tid == 0) {   // TDM: EXEC-independent, wave0 issues; stage 0
    tdm_load_2d(lds0, A + (size_t)row0 * K,
                (unsigned)K, (unsigned)(M - row0), BK, BM, (unsigned)K, 1);
    tdm_load_2d(lds0 + BM * AROW * 2, Bt + (size_t)col0 * K,
                (unsigned)K, (unsigned)(N - col0), BK, BN, (unsigned)K, 1);
  }
  for (int ks = 0; ks < nk; ++ks) {
    const int buf = ks & 1;
    if (ks + 1 < nk) {
      __syncthreads();                          // stage buf^1 free to overwrite
      if (tid == 0) {
        const int k1 = (ks + 1) * BK;
        tdm_load_2d(lds0 + (buf ^ 1) * STAGE_BYTES,
                    A + (size_t)row0 * K + k1,
                    (unsigned)(K - k1), (unsigned)(M - row0), BK, BM,
                    (unsigned)K, 1);
        tdm_load_2d(lds0 + (buf ^ 1) * STAGE_BYTES + BM * AROW * 2,
                    Bt + (size_t)col0 * K + k1,
                    (unsigned)(K - k1), (unsigned)(N - col0), BK, BN,
                    (unsigned)K, 1);
      }
    }
    if (tid == 0) {
      if (ks + 1 < nk) __builtin_amdgcn_s_wait_tensorcnt(2);  // stage ks done
      else             __builtin_amdgcn_s_wait_tensorcnt(0);
    }
    __syncthreads();
    const _Float16* As = &sm[buf * STAGE_H];
    const _Float16* Bs = As + BM * AROW;
    v16h af[2], bf[4];
#pragma unroll
    for (int i = 0; i < 2; ++i)
      af[i] = frag_ld(As + ((wm * 2 + i) * 16 + r) * AROW + hl * 8);
#pragma unroll
    for (int j = 0; j < 4; ++j)
      bf[j] = frag_ld(Bs + ((wn * 4 + j) * 16 + r) * AROW + hl * 8);
#pragma unroll
    for (int i = 0; i < 2; ++i)
#pragma unroll
      for (int j = 0; j < 4; ++j)
        acc[i][j] = wmma16(af[i], bf[j], acc[i][j]);
  }
  // epilogue: C/D layout -> row = e + 8*hl (+tile), col = lane&15 (+tile)
#pragma unroll
  for (int i = 0; i < 2; ++i) {
#pragma unroll
    for (int j = 0; j < 4; ++j) {
      const int n = col0 + (wn * 4 + j) * 16 + r;
      const float scl = colscale ? colscale[n] : 1.0f;
#pragma unroll
      for (int e = 0; e < 8; ++e) {
        const int m = row0 + (wm * 2 + i) * 16 + e + 8 * hl;
        if (m < M) {
          const float vv = acc[i][j][e] * scl;
          if (c_f32) ((float*)C)[(size_t)m * N + n] = vv;
          else       ((_Float16*)C)[(size_t)m * N + n] = (_Float16)vv;
        }
      }
    }
  }
}

// ---------------------------------------------------------- small prep kernels
__global__ void cvt_f16_kernel(const float* __restrict__ src,
                               _Float16* __restrict__ dst, long long n) {
  long long i = (long long)blockIdx.x * blockDim.x + threadIdx.x;
  const long long stride = (long long)gridDim.x * blockDim.x;
  for (; i < n; i += stride) dst[i] = (_Float16)src[i];
}

__global__ void cvt_pos_kernel(const float* __restrict__ src,
                               _Float16* __restrict__ dst, int nsrc, int ntot) {
  const int i = blockIdx.x * blockDim.x + threadIdx.x;
  if (i < ntot) dst[i] = (i < nsrc) ? (_Float16)src[i] : (_Float16)0.0f;
}

__global__ void qscale_kernel(const float* __restrict__ pds,
                              float* __restrict__ qs, int n, int d) {
  const int i = blockIdx.x * blockDim.x + threadIdx.x;
  if (i < n) {
    const float qsc = 1.4426950408889634f / sqrtf((float)d);
    qs[i] = qsc * log1pf(expf(pds[i % d]));    // Q_SCALE * softplus
  }
}

__global__ __launch_bounds__(256) void transpose_cvt(
    const float* __restrict__ W, _Float16* __restrict__ Wt, int K, int N) {
  __shared__ float tile[32][33];
  const int n0 = blockIdx.x * 32, k0 = blockIdx.y * 32;
  const int tx = threadIdx.x & 31, ty = threadIdx.x >> 5;
#pragma unroll
  for (int i = ty; i < 32; i += 8) tile[i][tx] = W[(size_t)(k0 + i) * N + n0 + tx];
  __syncthreads();
#pragma unroll
  for (int i = ty; i < 32; i += 8)
    Wt[(size_t)(n0 + i) * K + k0 + tx] = (_Float16)tile[tx][i];
}

// ------------------------------------------------------------ blocked attention
// One wave per (b, block, head). 128 threads handle 4 heads.
__global__ __launch_bounds__(128) void attn_kernel(
    const _Float16* __restrict__ qh, const _Float16* __restrict__ kh,
    const _Float16* __restrict__ vh, const _Float16* __restrict__ relh,
    _Float16* __restrict__ oh, int S, int NB) {
  __shared__ _Float16 vt[4][kD * 32];          // V transposed: [d][ctx]
  __shared__ float    sc[4][16 * 32];          // scores, later reused as p(f16)
  __shared__ float    bdl[4][16 * 16];         // rel scores

  const int blk = blockIdx.x;
  const int b  = blk / (NB * 2);
  const int rm = blk % (NB * 2);
  const int nb = rm >> 1;
  const int hh = rm & 1;
  const int w = threadIdx.x >> 5, lane = threadIdx.x & 31;
  const int h = hh * 4 + w;
  const int hl = lane >> 4, r = lane & 15;

  _Float16* VT = vt[w];
  float* SC = sc[w];
  float* BD = bdl[w];

  { // zero V tile (covers invalid ctx rows and ctx 24..31 padding)
    unsigned* z = (unsigned*)VT;
    for (int i = lane; i < kD * 32 / 2; i += 32) z[i] = 0u;
  }
  asm volatile("" ::: "memory");
  // gather V rows, write transposed [d][c]
  for (int c = 0; c < kCTX; ++c) {
    const int ak = nb * kCHUNK - 12 + c;
    if (ak < 0 || ak >= S) continue;
    const unsigned* vr =
        (const unsigned*)(vh + ((size_t)b * S + ak) * kHID + h * kD);
#pragma unroll
    for (int t = 0; t < 3; ++t) {
      const int i = t * 32 + lane;
      const unsigned u = vr[i];
      const int d = 2 * i;
      VT[d * 32 + c]       = ((const _Float16*)&u)[0];
      VT[(d + 1) * 32 + c] = ((const _Float16*)&u)[1];
    }
  }

  // ---- QK^T + rel scores (A = q rows, B = k rows / rel rows)
  const int aq = nb * kCHUNK + r;
  const bool aok = (r < kCHUNK) && (aq < S);
  const _Float16* qrow = qh + ((size_t)b * S + (aok ? aq : 0)) * kHID + h * kD;

  const int ak0 = nb * kCHUNK - 12 + r;
  const bool k0ok = (ak0 >= 0) && (ak0 < S);
  const _Float16* krow0 = kh + ((size_t)b * S + (k0ok ? ak0 : 0)) * kHID + h * kD;

  const int c1 = 16 + r;
  const int ak1 = nb * kCHUNK - 12 + c1;
  const bool k1ok = (c1 < kCTX) && (ak1 >= 0) && (ak1 < S);
  const _Float16* krow1 = kh + ((size_t)b * S + (k1ok ? ak1 : 0)) * kHID + h * kD;

  const _Float16* rrow = relh + (size_t)r * kHID + h * kD;  // rows 13..15 are 0

  v8f a0, a1, ab;
#pragma unroll
  for (int e = 0; e < 8; ++e) { a0[e] = 0.f; a1[e] = 0.f; ab[e] = 0.f; }
  const v16h fz = frag_zero();
#pragma unroll
  for (int ks = 0; ks < kD / 32; ++ks) {
    const int off = ks * 32 + hl * 8;
    const v16h qa  = aok  ? frag_ld(qrow + off)  : fz;
    const v16h kb0 = k0ok ? frag_ld(krow0 + off) : fz;
    const v16h kb1 = k1ok ? frag_ld(krow1 + off) : fz;
    const v16h rb  = frag_ld(rrow + off);
    a0 = wmma16(qa, kb0, a0);
    a1 = wmma16(qa, kb1, a1);
    ab = wmma16(qa, rb, ab);
  }
#pragma unroll
  for (int e = 0; e < 8; ++e) {
    const int m = e + 8 * hl;
    SC[m * 32 + r]      = a0[e];
    SC[m * 32 + 16 + r] = a1[e];
    BD[m * 16 + r]      = ab[e];
  }
  asm volatile("s_wait_dscnt 0" ::: "memory");

  // ---- rel-shift + softcap + mask + softmax; lane <-> query row
  float prob[kCTX];
  const int m = lane;
  const int aqm = nb * kCHUNK + m;
  const bool rok = (m < kCHUNK) && (aqm < S);
  if (rok) {
    float mx = -1e30f;
#pragma unroll
    for (int c = 0; c < kCTX; ++c) {
      float s = SC[m * 32 + c];
      const int p = c - m;                      // rel-shift: bd[q][c-q]
      if (p >= 0 && p < kPL) s += BD[m * 16 + p];
      s = tanhf(s * (1.0f / 50.0f)) * 50.0f;    // softcap
      const int ak = nb * kCHUNK - 12 + c;
      const bool keep = (c > m) && (c <= m + 12) && (ak >= 0) && (ak < S);
      s = keep ? s : -1e9f;
      prob[c] = s;
      mx = fmaxf(mx, s);
    }
    float sum = 0.0f;
#pragma unroll
    for (int c = 0; c < kCTX; ++c) {
      const float e2 = expf(prob[c] - mx);
      prob[c] = e2; sum += e2;
    }
    const float inv = 1.0f / sum;
#pragma unroll
    for (int c = 0; c < kCTX; ++c) prob[c] *= inv;
  }
  asm volatile("" ::: "memory");
  // overlay p (f16) onto SC
  _Float16* P16 = (_Float16*)SC;
  {
    unsigned* z = (unsigned*)P16;
    for (int i = lane; i < 16 * 32 / 2; i += 32) z[i] = 0u;
  }
  asm volatile("" ::: "memory");
  if (rok) {
#pragma unroll
    for (int c = 0; c < kCTX; ++c) P16[m * 32 + c] = (_Float16)prob[c];
  }
  asm volatile("s_wait_dscnt 0" ::: "memory");

  // ---- out = p @ vc : A = p [16x32], B = V^T columns from LDS, 12 N-tiles
  const v16h pf = frag_ld(P16 + r * 32 + hl * 8);
#pragma unroll
  for (int tn = 0; tn < kD / 16; ++tn) {
    const v16h vb = frag_ld(VT + (tn * 16 + r) * 32 + hl * 8);
    v8f o;
#pragma unroll
    for (int e = 0; e < 8; ++e) o[e] = 0.0f;
    o = wmma16(pf, vb, o);
#pragma unroll
    for (int e = 0; e < 8; ++e) {
      const int mm = e + 8 * hl;
      const int aqo = nb * kCHUNK + mm;
      if (mm < kCHUNK && aqo < S)
        oh[((size_t)b * S + aqo) * kHID + h * kD + tn * 16 + r] = (_Float16)o[e];
    }
  }
}

// ------------------------------------------------------------------- launcher
extern "C" void kernel_launch(void* const* d_in, const int* in_sizes, int n_in,
                              void* d_out, int out_size, void* d_ws,
                              size_t ws_size, hipStream_t stream) {
  (void)n_in; (void)out_size; (void)ws_size;
  const float* x     = (const float*)d_in[0];
  const float* pos   = (const float*)d_in[1];
  const float* Wq    = (const float*)d_in[2];
  const float* Wk    = (const float*)d_in[3];
  const float* Wv    = (const float*)d_in[4];
  const float* Wrel  = (const float*)d_in[5];
  const float* pds   = (const float*)d_in[6];
  const float* Wpost = (const float*)d_in[7];

  const int S  = in_sizes[0] / (kB * kHID);   // 6007
  const int M  = kB * S;                      // 24028
  const int NB = (S + kCHUNK - 1) / kCHUNK;   // 501

  char* ws = (char*)d_ws;
  size_t off = 0;
  auto carve = [&](size_t bytes) -> char* {
    char* p = ws + off;
    off += (bytes + 255) & ~(size_t)255;
    return p;
  };
  const size_t big = (size_t)M * kHID * sizeof(_Float16);
  const size_t wsz = (size_t)kHID * kHID * sizeof(_Float16);
  _Float16* xh   = (_Float16*)carve(big);     // x in f16; later reused as attn out
  _Float16* q16  = (_Float16*)carve(big);
  _Float16* k16  = (_Float16*)carve(big);
  _Float16* v16b = (_Float16*)carve(big);
  _Float16* Wqt  = (_Float16*)carve(wsz);
  _Float16* Wkt  = (_Float16*)carve(wsz);
  _Float16* Wvt  = (_Float16*)carve(wsz);
  _Float16* Wrt  = (_Float16*)carve(wsz);
  _Float16* Wpt  = (_Float16*)carve(wsz);
  _Float16* ph   = (_Float16*)carve((size_t)16 * kHID * sizeof(_Float16));
  _Float16* rel16= (_Float16*)carve((size_t)16 * kHID * sizeof(_Float16));
  float*    qs   = (float*)carve((size_t)kHID * sizeof(float));

  const long long nx = (long long)M * kHID;
  cvt_f16_kernel<<<2048, 256, 0, stream>>>(x, xh, nx);
  cvt_pos_kernel<<<(16 * kHID + 255) / 256, 256, 0, stream>>>(pos, ph,
                                                              kPL * kHID,
                                                              16 * kHID);
  qscale_kernel<<<(kHID + 255) / 256, 256, 0, stream>>>(pds, qs, kHID, kD);
  dim3 tg(kHID / 32, kHID / 32);
  transpose_cvt<<<tg, 256, 0, stream>>>(Wq,    Wqt, kHID, kHID);
  transpose_cvt<<<tg, 256, 0, stream>>>(Wk,    Wkt, kHID, kHID);
  transpose_cvt<<<tg, 256, 0, stream>>>(Wv,    Wvt, kHID, kHID);
  transpose_cvt<<<tg, 256, 0, stream>>>(Wrel,  Wrt, kHID, kHID);
  transpose_cvt<<<tg, 256, 0, stream>>>(Wpost, Wpt, kHID, kHID);

  dim3 gg(kHID / BN, (M + BM - 1) / BM);
  gemm_f16_wmma<<<gg, 256, 0, stream>>>(xh, Wqt, q16, qs,      M, kHID, kHID, 0);
  gemm_f16_wmma<<<gg, 256, 0, stream>>>(xh, Wkt, k16, nullptr, M, kHID, kHID, 0);
  gemm_f16_wmma<<<gg, 256, 0, stream>>>(xh, Wvt, v16b, nullptr, M, kHID, kHID, 0);
  dim3 gr(kHID / BN, 1);
  gemm_f16_wmma<<<gr, 256, 0, stream>>>(ph, Wrt, rel16, nullptr, 16, kHID, kHID, 0);

  attn_kernel<<<kB * NB * 2, 128, 0, stream>>>(q16, k16, v16b, rel16, xh, S, NB);

  gemm_f16_wmma<<<gg, 256, 0, stream>>>(xh, Wpt, d_out, nullptr, M, kHID, kHID, 1);
}
